// AMRTransformer_80367428043292
// MI455X (gfx1250) — matrix operand order repaired and checked by reading, not compiled, not run on hardware
//
#include <hip/hip_runtime.h>
#include <hip/hip_bf16.h>

// Problem constants (match reference)
#define NUM_GRAPHS 64
#define EDGES_PER_GRAPH 4096
#define NN 16384          // nodes
#define EE 262144         // edges
#define DD 256            // model dim
#define HH 8              // heads
#define LL 2              // layers

typedef __bf16 bf16_t;
typedef __attribute__((ext_vector_type(16))) __bf16 v16bf;
typedef __attribute__((ext_vector_type(8)))  __bf16 v8bf;
typedef __attribute__((ext_vector_type(8)))  float  v8f;

// ---------------------------------------------------------------------------
// Generic bf16 WMMA GEMM:  C[M,N] = gatherA[M,K] @ Bt^T   (Bt is [N,K] bf16)
// A may be split into two sources (concat along K) with optional row-gather
// indices. Epilogue: optional bias (f32 [N]), optional ReLU, writes f32
// and/or bf16 outputs. One wave computes a 16(M) x 128(N) tile (8 acc) so A
// fragments are reused 8x. All fragment loads of a K-chunk are issued before
// a sched_barrier so they clause together and retire against staggered
// s_wait_loadcnt, overlapping later tiles' loads with earlier tiles' WMMAs
// (the scheduler otherwise sinks each load to just before its WMMA).
// ---------------------------------------------------------------------------
__global__ __launch_bounds__(256)
void gemm_bf16_wmma_kernel(const unsigned short* __restrict__ A1u, int lda1,
                           const unsigned short* __restrict__ A2u, int lda2,
                           const int* __restrict__ idx1,
                           const int* __restrict__ idx2,
                           const unsigned short* __restrict__ Btu,
                           float* __restrict__ Cf,
                           unsigned short* __restrict__ Cbfu,
                           const float* __restrict__ bias,
                           int M, int N, int K, int relu)
{
    const bf16_t* A1  = reinterpret_cast<const bf16_t*>(A1u);
    const bf16_t* A2  = reinterpret_cast<const bf16_t*>(A2u);
    const bf16_t* Bt  = reinterpret_cast<const bf16_t*>(Btu);
    bf16_t*       Cbf = reinterpret_cast<bf16_t*>(Cbfu);

    const int lane   = threadIdx.x & 31;
    const int wave   = threadIdx.x >> 5;
    const int tilesN = N >> 7;                 // 128-wide N tiles per wave
    const int Mtiles = M >> 4;
    const int waveId = blockIdx.x * 8 + wave;
    const int tM     = waveId / tilesN;
    if (tM >= Mtiles) return;                  // wave-uniform exit (EXEC stays all-1)
    const int tN     = (waveId % tilesN) << 7;

    const int mrow = (tM << 4) + (lane & 15);
    const int kb   = (lane >> 4) << 4;         // A K-half:    0 or 16
    const int kseg = (lane >> 4) << 4;         // B K-half:    0 or 16

    const size_t r1 = (size_t)(idx1 ? idx1[mrow] : mrow);
    const bf16_t* a1p = A1 + r1 * (size_t)lda1;
    const bf16_t* a2p = nullptr;
    if (A2u) {
        const size_t r2 = (size_t)(idx2 ? idx2[mrow] : mrow);
        a2p = A2 + r2 * (size_t)lda2;
    }

    v8f acc[8];
#pragma unroll
    for (int j = 0; j < 8; ++j)
#pragma unroll
        for (int r = 0; r < 8; ++r) acc[j][r] = 0.0f;

    // Per-tile B row pointers (column n = colBase + 16j of Bt), K-half offset.
    const int colBase = tN + (lane & 15);
    const bf16_t* bp[8];
#pragma unroll
    for (int j = 0; j < 8; ++j)
        bp[j] = Bt + (size_t)(colBase + (j << 4)) * (size_t)K + (size_t)kseg;

    for (int kk = 0; kk < K; kk += 32) {
        const bf16_t* ap; int kl;
        if (a2p && kk >= lda1) { ap = a2p; kl = kk - lda1; }   // concat-K second half
        else                   { ap = a1p; kl = kk; }
        union { v16bf v; v8bf h[2]; } af;
        af.h[0] = *(const v8bf*)(ap + kl + kb);
        af.h[1] = *(const v8bf*)(ap + kl + kb + 8);

        // Batch all B fragment loads; the sched barrier keeps them issued
        // ahead of the WMMA chain (8 live fragments, staggered waits).
        v16bf bfrag[8];
#pragma unroll
        for (int j = 0; j < 8; ++j)
            bfrag[j] = *(const v16bf*)(bp[j] + kk);
        __builtin_amdgcn_sched_barrier(0);
#pragma unroll
        for (int j = 0; j < 8; ++j)
            acc[j] = __builtin_amdgcn_wmma_f32_16x16x32_bf16(
                         false, af.v, false, bfrag[j], (short)0, acc[j], false, false);
    }

    // C layout: lane n = lane&15 ; VGPR r holds row (tM*16 + 8*(lane>>4) + r)
    const int rbase = (tM << 4) + ((lane >> 4) << 3);
#pragma unroll
    for (int j = 0; j < 8; ++j) {
        const int n = tN + (j << 4) + (lane & 15);
        const float bvv = bias ? bias[n] : 0.0f;
#pragma unroll
        for (int r = 0; r < 8; ++r) {
            float v = acc[j][r] + bvv;
            if (relu) v = fmaxf(v, 0.0f);
            const size_t off = (size_t)(rbase + r) * (size_t)N + (size_t)n;
            if (Cf)   Cf[off]  = v;
            if (Cbfu) Cbf[off] = (bf16_t)v;
        }
    }
}

// ---------------------------------------------------------------------------
// Token embedding gather -> xs/xt f32 + bf16 mirrors
// ---------------------------------------------------------------------------
__global__ __launch_bounds__(256)
void embed_kernel(const int* __restrict__ tok, const float* __restrict__ emb,
                  float* __restrict__ xs, float* __restrict__ xt,
                  unsigned short* __restrict__ xsbu, unsigned short* __restrict__ xtbu)
{
    bf16_t* xsb = reinterpret_cast<bf16_t*>(xsbu);
    bf16_t* xtb = reinterpret_cast<bf16_t*>(xtbu);
    int i = blockIdx.x * 256 + threadIdx.x;
    if (i >= NN * DD) return;
    int n = i >> 8, d = i & 255;
    float v = emb[(size_t)tok[n] * DD + d];
    xs[i] = v; xt[i] = v;
    xsb[i] = (bf16_t)v; xtb[i] = (bf16_t)v;
}

// Weight prep: f32 [K,N] row-major -> bf16 [N,K] (transposed for B-fragment loads)
__global__ __launch_bounds__(256)
void wtrans_kernel(const float* __restrict__ src, unsigned short* __restrict__ dstu,
                   int K, int N)
{
    bf16_t* dst = reinterpret_cast<bf16_t*>(dstu);
    int i = blockIdx.x * 256 + threadIdx.x;
    if (i >= K * N) return;
    int k = i / N, n = i - k * N;
    dst[(size_t)n * K + k] = (bf16_t)src[i];
}

__global__ __launch_bounds__(256)
void f2b_kernel(const float* __restrict__ s, unsigned short* __restrict__ du, int n)
{
    bf16_t* d = reinterpret_cast<bf16_t*>(du);
    int i = blockIdx.x * 256 + threadIdx.x;
    if (i < n) d[i] = (bf16_t)s[i];
}

// ---------------------------------------------------------------------------
// Attention logits: a[e,h] = scale * <Q[dst[e],h,:], K[e,h,:]>  (hd = 32)
// One wave per edge; lane l handles dims [8l,8l+8); 4-lane xor-reduce per head.
// ---------------------------------------------------------------------------
__global__ __launch_bounds__(256)
void logits_kernel(const float* __restrict__ Q, const unsigned short* __restrict__ Kbu,
                   const int* __restrict__ dst, float* __restrict__ a, int E)
{
    const bf16_t* Kb = reinterpret_cast<const bf16_t*>(Kbu);
    int e = blockIdx.x * 8 + (threadIdx.x >> 5);
    if (e >= E) return;
    int lane = threadIdx.x & 31;
    size_t qoff = (size_t)dst[e] * DD + lane * 8;
    size_t koff = (size_t)e * DD + lane * 8;
    float s = 0.0f;
#pragma unroll
    for (int j = 0; j < 8; ++j) s += Q[qoff + j] * (float)Kb[koff + j];
    s += __shfl_xor(s, 1);
    s += __shfl_xor(s, 2);
    if ((lane & 3) == 0)
        a[(size_t)e * HH + (lane >> 2)] = s * 0.17677669529663687f;  // 32^-0.5
}

// Per-(graph,head) softmax over the contiguous 4096-edge segment, in place.
__global__ __launch_bounds__(256)
void segsoftmax_kernel(float* __restrict__ a)
{
    __shared__ float red[256];
    int g = blockIdx.x >> 3;
    int h = blockIdx.x & 7;
    size_t base = (size_t)g * EDGES_PER_GRAPH;
    int t = threadIdx.x;

    float m = -1e30f;
    for (int i = t; i < EDGES_PER_GRAPH; i += 256)
        m = fmaxf(m, a[(base + i) * HH + h]);
    red[t] = m; __syncthreads();
    for (int s = 128; s > 0; s >>= 1) { if (t < s) red[t] = fmaxf(red[t], red[t + s]); __syncthreads(); }
    m = red[0]; __syncthreads();

    float sum = 0.0f;
    for (int i = t; i < EDGES_PER_GRAPH; i += 256) {
        size_t o = (base + i) * HH + h;
        float e = __expf(a[o] - m);
        a[o] = e; sum += e;
    }
    red[t] = sum; __syncthreads();
    for (int s = 128; s > 0; s >>= 1) { if (t < s) red[t] += red[t + s]; __syncthreads(); }
    float inv = 1.0f / red[0];
    for (int i = t; i < EDGES_PER_GRAPH; i += 256)
        a[(base + i) * HH + h] *= inv;
}

// ---------------------------------------------------------------------------
// Segmented scatter via LDS: one block per (graph, head-slice). Head h owns
// dims [32h, 32h+32) which all share softmax weight a[e,h]. Accumulator is
// 256 nodes x 32 dims f32 (32 KB LDS). 32 lanes of a wave hit 32 consecutive
// banks -> conflict-free ds_add_f32. agg is fully covered, no pre-zero needed.
// ---------------------------------------------------------------------------
__global__ __launch_bounds__(256)
void scatter_lds_kernel(const unsigned short* __restrict__ Vbu,
                        const float* __restrict__ a,
                        const int* __restrict__ dst,
                        float* __restrict__ agg)
{
    __shared__ float acc[256 * 32];
    const bf16_t* Vb = reinterpret_cast<const bf16_t*>(Vbu);
    const int g = blockIdx.x >> 3;      // graph
    const int s = blockIdx.x & 7;       // head / 32-dim slice
    const int t = threadIdx.x;

    for (int i = t; i < 256 * 32; i += 256) acc[i] = 0.0f;
    __syncthreads();

    const int d  = t & 31;
    const int eo = t >> 5;              // 8 edge lanes (one per wave)
    const size_t ebase = (size_t)g * EDGES_PER_GRAPH;
    for (int e = eo; e < EDGES_PER_GRAPH; e += 8) {
        const size_t ge = ebase + e;
        const float w = a[ge * HH + s];
        const int   nl = dst[ge] & 255;                 // node local to graph
        const float v = (float)Vb[ge * DD + s * 32 + d] * w;
        atomicAdd(&acc[nl * 32 + d], v);
    }
    __syncthreads();

    for (int i = t; i < 256 * 32; i += 256) {
        const int nl = i >> 5, dd = i & 31;
        agg[((size_t)g * 256 + nl) * DD + s * 32 + dd] = acc[i];
    }
}

// gate = sigmoid(glin); out_bf = bf16(gate*out_s + (1-gate)*out_t)
__global__ __launch_bounds__(256)
void gate_kernel(const float* __restrict__ glin, const float* __restrict__ os,
                 const float* __restrict__ ot, unsigned short* __restrict__ obu, int n)
{
    bf16_t* ob = reinterpret_cast<bf16_t*>(obu);
    int i = blockIdx.x * 256 + threadIdx.x;
    if (i >= n) return;
    float gv = 1.0f / (1.0f + __expf(-glin[i]));
    ob[i] = (bf16_t)(gv * os[i] + (1.0f - gv) * ot[i]);
}

// LayerNorm(x + ff2[:, half*256 : half*256+256]) in place; also bf16 mirror.
__global__ __launch_bounds__(256)
void ln_kernel(float* __restrict__ x, unsigned short* __restrict__ xbu,
               const float* __restrict__ ff2, int half,
               const float* __restrict__ g, const float* __restrict__ b)
{
    bf16_t* xb = reinterpret_cast<bf16_t*>(xbu);
    __shared__ float red[256];
    int n = blockIdx.x;
    int d = threadIdx.x;
    float v = x[(size_t)n * DD + d] + ff2[(size_t)n * 512 + half * 256 + d];
    red[d] = v; __syncthreads();
    for (int s = 128; s > 0; s >>= 1) { if (d < s) red[d] += red[d + s]; __syncthreads(); }
    float mu = red[0] * (1.0f / 256.0f); __syncthreads();
    float c = v - mu;
    red[d] = c * c; __syncthreads();
    for (int s = 128; s > 0; s >>= 1) { if (d < s) red[d] += red[d + s]; __syncthreads(); }
    float var = red[0] * (1.0f / 256.0f);
    float y = c * rsqrtf(var + 1e-5f) * g[d] + b[d];
    x[(size_t)n * DD + d] = y;
    xb[(size_t)n * DD + d] = (bf16_t)y;
}

__global__ __launch_bounds__(256)
void concat_kernel(const float* __restrict__ xs, const float* __restrict__ xt,
                   float* __restrict__ out)
{
    int i = blockIdx.x * 256 + threadIdx.x;
    if (i >= NN * 512) return;
    int n = i >> 9, d = i & 511;
    out[i] = (d < 256) ? xs[(size_t)n * DD + d] : xt[(size_t)n * DD + (d - 256)];
}

// ---------------------------------------------------------------------------
extern "C" void kernel_launch(void* const* d_in, const int* in_sizes, int n_in,
                              void* d_out, int out_size, void* d_ws, size_t ws_size,
                              hipStream_t stream)
{
    (void)in_sizes; (void)n_in; (void)out_size; (void)ws_size;
    const int*   tok = (const int*)d_in[0];
    const int*   e0  = (const int*)d_in[1];
    const int*   e1  = (const int*)d_in[2];
    /* d_in[3] = edge_graph: edges are contiguous per graph, not needed */
    const float* emb = (const float*)d_in[4];
    const float* Wr  = (const float*)d_in[5];
    const float* Wq  = (const float*)d_in[6];
    const float* Wk  = (const float*)d_in[7];
    const float* Wv  = (const float*)d_in[8];
    const float* Wc  = (const float*)d_in[9];
    const float* W1  = (const float*)d_in[10];
    const float* W2  = (const float*)d_in[11];
    const float* b2  = (const float*)d_in[12];
    const float* Wo  = (const float*)d_in[13];
    const float* bo  = (const float*)d_in[14];
    const float* lng = (const float*)d_in[15];
    const float* lnb = (const float*)d_in[16];
    float* out = (float*)d_out;

    typedef unsigned short u16;
    char* ws = (char*)d_ws;
    size_t cur = 0;
    auto alloc = [&](size_t bytes) -> char* {
        char* p = ws + cur;
        cur += (bytes + 255) & ~(size_t)255;
        return p;
    };
    const size_t ND = (size_t)NN * DD, ED = (size_t)EE * DD;
    float* xs    = (float*)alloc(ND * 4);
    float* xt    = (float*)alloc(ND * 4);
    u16*   xsb   = (u16*)  alloc(ND * 2);
    u16*   xtb   = (u16*)  alloc(ND * 2);
    float* Qs    = (float*)alloc(ND * 4);
    float* Qt    = (float*)alloc(ND * 4);
    u16*   Rb    = (u16*)  alloc(ED * 2);
    u16*   Kb    = (u16*)  alloc(ED * 2);
    u16*   Vb    = (u16*)  alloc(ED * 2);
    float* att   = (float*)alloc((size_t)EE * HH * 4);
    float* agg   = (float*)alloc(ND * 4);
    u16*   aggb  = (u16*)  alloc(ND * 2);
    float* outs  = (float*)alloc(ND * 4);
    u16*   outsb = (u16*)  alloc(ND * 2);
    float* outt  = (float*)alloc(ND * 4);
    u16*   outtb = (u16*)  alloc(ND * 2);
    float* glin  = (float*)alloc(ND * 4);
    u16*   outb  = (u16*)  alloc(ND * 2);
    u16*   ff1b  = (u16*)  alloc((size_t)NN * 1024 * 2);
    float* ff2   = (float*)alloc((size_t)NN * 512 * 4);
    u16*   WrT   = (u16*)  alloc((size_t)LL * 512 * 256 * 2);
    u16*   WqT   = (u16*)  alloc((size_t)LL * 256 * 256 * 2);
    u16*   WkT   = (u16*)  alloc((size_t)LL * 256 * 256 * 2);
    u16*   WvT   = (u16*)  alloc((size_t)LL * 256 * 256 * 2);
    u16*   WcT   = (u16*)  alloc((size_t)LL * 512 * 256 * 2);
    u16*   W1T   = (u16*)  alloc((size_t)LL * 256 * 1024 * 2);
    u16*   W2T   = (u16*)  alloc((size_t)LL * 1024 * 512 * 2);
    u16*   WoT   = (u16*)  alloc((size_t)LL * 256 * 256 * 2);

    auto gemm = [&](const u16* A1, int lda1, const int* i1,
                    const u16* A2, int lda2, const int* i2,
                    const u16* Bt, float* Cf, u16* Cbf, const float* bias,
                    int M, int N, int K, int relu) {
        int tiles  = (M / 16) * (N / 128);
        int blocks = (tiles + 7) / 8;
        gemm_bf16_wmma_kernel<<<blocks, 256, 0, stream>>>(
            A1, lda1, A2, lda2, i1, i2, Bt, Cf, Cbf, bias, M, N, K, relu);
    };
    auto wtrans = [&](const float* src, u16* dst, int K, int N) {
        wtrans_kernel<<<(K * N + 255) / 256, 256, 0, stream>>>(src, dst, K, N);
    };

    // --- prep ---
    embed_kernel<<<(int)(ND / 256), 256, 0, stream>>>(tok, emb, xs, xt, xsb, xtb);
    for (int l = 0; l < LL; ++l) {
        wtrans(Wr + (size_t)l * 512 * 256, WrT + (size_t)l * 512 * 256, 512, 256);
        wtrans(Wq + (size_t)l * 256 * 256, WqT + (size_t)l * 256 * 256, 256, 256);
        wtrans(Wk + (size_t)l * 256 * 256, WkT + (size_t)l * 256 * 256, 256, 256);
        wtrans(Wv + (size_t)l * 256 * 256, WvT + (size_t)l * 256 * 256, 256, 256);
        wtrans(Wc + (size_t)l * 512 * 256, WcT + (size_t)l * 512 * 256, 512, 256);
        wtrans(W1 + (size_t)l * 256 * 1024, W1T + (size_t)l * 256 * 1024, 256, 1024);
        wtrans(W2 + (size_t)l * 1024 * 512, W2T + (size_t)l * 1024 * 512, 1024, 512);
        wtrans(Wo + (size_t)l * 256 * 256, WoT + (size_t)l * 256 * 256, 256, 256);
    }

    for (int l = 0; l < LL; ++l) {
        const u16* wrt = WrT + (size_t)l * 512 * 256;
        const u16* wqt = WqT + (size_t)l * 256 * 256;
        const u16* wkt = WkT + (size_t)l * 256 * 256;
        const u16* wvt = WvT + (size_t)l * 256 * 256;
        const u16* wct = WcT + (size_t)l * 512 * 256;
        const u16* w1t = W1T + (size_t)l * 256 * 1024;
        const u16* w2t = W2T + (size_t)l * 1024 * 512;
        const u16* wot = WoT + (size_t)l * 256 * 256;
        const float* bol = bo + (size_t)l * 256;
        const float* b2l = b2 + (size_t)l * 512;

        // R = [xs[e0], xt[e1]] @ Wr   (gathered, concat-K fused)  -> bf16
        gemm(xsb, 256, e0, xtb, 256, e1, wrt, nullptr, Rb, nullptr, EE, 256, 512, 0);
        // Qs, Qt
        gemm(xsb, 256, nullptr, nullptr, 0, nullptr, wqt, Qs, nullptr, nullptr, NN, 256, 256, 0);
        gemm(xtb, 256, nullptr, nullptr, 0, nullptr, wqt, Qt, nullptr, nullptr, NN, 256, 256, 0);
        // V = R @ Wv, K = R @ Wk  -> bf16
        gemm(Rb, 256, nullptr, nullptr, 0, nullptr, wvt, nullptr, Vb, nullptr, EE, 256, 256, 0);
        gemm(Rb, 256, nullptr, nullptr, 0, nullptr, wkt, nullptr, Kb, nullptr, EE, 256, 256, 0);

        // ---- attend(Qs, e0) -> out_s ----
        logits_kernel<<<EE / 8, 256, 0, stream>>>(Qs, Kb, e0, att, EE);
        segsoftmax_kernel<<<NUM_GRAPHS * HH, 256, 0, stream>>>(att);
        scatter_lds_kernel<<<NUM_GRAPHS * 8, 256, 0, stream>>>(Vb, att, e0, agg);
        f2b_kernel<<<(int)(ND / 256), 256, 0, stream>>>(agg, aggb, (int)ND);
        gemm(aggb, 256, nullptr, nullptr, 0, nullptr, wot, outs, outsb, bol, NN, 256, 256, 0);

        // ---- attend(Qt, e1) -> out_t ----
        logits_kernel<<<EE / 8, 256, 0, stream>>>(Qt, Kb, e1, att, EE);
        segsoftmax_kernel<<<NUM_GRAPHS * HH, 256, 0, stream>>>(att);
        scatter_lds_kernel<<<NUM_GRAPHS * 8, 256, 0, stream>>>(Vb, att, e1, agg);
        f2b_kernel<<<(int)(ND / 256), 256, 0, stream>>>(agg, aggb, (int)ND);
        gemm(aggb, 256, nullptr, nullptr, 0, nullptr, wot, outt, outtb, bol, NN, 256, 256, 0);

        // gate = sigmoid([out_s, out_t] @ Wc); out = gate*out_s + (1-gate)*out_t
        gemm(outsb, 256, nullptr, outtb, 256, nullptr, wct, glin, nullptr, nullptr, NN, 256, 512, 0);
        gate_kernel<<<(int)(ND / 256), 256, 0, stream>>>(glin, outs, outt, outb, (int)ND);

        // ff = relu(out @ W1) @ W2 + b2
        gemm(outb, 256, nullptr, nullptr, 0, nullptr, w1t, nullptr, ff1b, nullptr, NN, 1024, 256, 1);
        gemm(ff1b, 1024, nullptr, nullptr, 0, nullptr, w2t, ff2, nullptr, b2l, NN, 512, 1024, 0);

        // residual + layernorm -> xs, xt (f32 + bf16 mirrors)
        ln_kernel<<<NN, 256, 0, stream>>>(xs, xsb, ff2, 0, lng + (size_t)l * 256, lnb + (size_t)l * 256);
        ln_kernel<<<NN, 256, 0, stream>>>(xt, xtb, ff2, 1, lng + (size_t)l * 256, lnb + (size_t)l * 256);
    }

    concat_kernel<<<(NN * 512 + 255) / 256, 256, 0, stream>>>(xs, xt, out);
}
